// KoopmanModel_23759759081742
// MI455X (gfx1250) — matrix operand order repaired and checked by reading, not compiled, not run on hardware
//
#include <hip/hip_runtime.h>
#include <math.h>

// ---------------------------------------------------------------------------
// Performer (FAVOR+) cross-attention block for MI455X (gfx1250, wave32, WMMA).
// All GEMMs use v_wmma_f32_16x16x32_f16. Memory-bound on the 453MB outputs
// (23.3 TB/s HBM => ~25us floor); WMMA f16 gives ample compute headroom for
// the ~20 GFLOP of GEMM work. Weights are pre-converted/pre-transposed to f16
// [k][n] layout once, so B fragments load f16 directly, each lane 16 values
// strided by 128 along k (no per-tile v_cvt). V is stored f16 for the same
// reason. Workspace use: ~144 MB.
// ---------------------------------------------------------------------------

typedef __attribute__((ext_vector_type(16))) _Float16 v16h;
typedef __attribute__((ext_vector_type(8)))  float    v8f;

#define WMMA_F16(a, b, c) \
  __builtin_amdgcn_wmma_f32_16x16x32_f16(false, (a), false, (b), (short)0, (c), false, false)

static constexpr int H = 8, DH = 16, DP = 128, DE = 32, M = 64;
static constexpr int Bb = 16, NCTX = 3200, NQ = 3072, S = 128;
static constexpr int B2 = 32;          // 2*B
static constexpr int BH = 256;         // 2*B*H
static constexpr float DN     = 0.5f;          // 16^-0.25
static constexpr float RATIO  = 0.125f;        // 64^-0.5
static constexpr float REPS   = 0.125f * 1e-4f;
static constexpr float DIAG_C = 0.125f;        // 0.5*dn*dn

// K index inside a 16x32 f16 A fragment for VGPR-half j, k-half kh (ISA 7.12.2)
__device__ __forceinline__ int a_kidx(int j, int kh) {
  return ((j & 4) ? 16 : 0) + kh * 8 + (j & 3) * 2;
}

// A fragment (16x32) from f16 tile in LDS, row stride ld, starting column k0.
__device__ __forceinline__ v16h ldsA_frag(const _Float16* t, int ld, int k0, int lane) {
  int m = lane & 15, kh = lane >> 4;
  v16h a;
#pragma unroll
  for (int j = 0; j < 8; ++j) {
    int k = k0 + a_kidx(j, kh);
    a[2 * j]     = t[m * ld + k];
    a[2 * j + 1] = t[m * ld + k + 1];
  }
  return a;
}

// B fragment (32x16) from pre-converted f16 weights stored [k][n], ldw = 128.
// w points at [k0][n0]; lane%16 = n, elements over k in a_kidx order.
__device__ __forceinline__ v16h gB16(const _Float16* w, int lane) {
  int n = lane & 15, kh = lane >> 4;
  v16h b;
#pragma unroll
  for (int j = 0; j < 8; ++j) {
    int k = a_kidx(j, kh);
    b[2 * j]     = w[(size_t)k * DP + n];
    b[2 * j + 1] = w[(size_t)(k + 1) * DP + n];
  }
  return b;
}

__device__ __forceinline__ float gelu_f(float x) {
  float t = 0.7978845608028654f * (x + 0.044715f * x * x * x);
  return 0.5f * x * (1.0f + tanhf(t));
}

__device__ __forceinline__ void atomicMaxF(float* p, float v) {
  unsigned int* u = (unsigned int*)p;
  unsigned int cur = __float_as_uint(*p);
  while (v > __uint_as_float(cur)) {
    unsigned int old = atomicCAS(u, cur, __float_as_uint(v));
    if (old == cur) break;
    cur = old;
  }
}

// ---------------------------------------------------------------------------
__global__ void k_init(float* kmax) {
  if (threadIdx.x < BH) kmax[threadIdx.x] = -__builtin_inff();
}

// ---------------------------------------------------------------------------
// One-shot weight conversion to f16, all in [k][n] layout with ldw = 128:
//   fcw16 = fcw, prw16 = prw (already [k][n])
//   wq16/wo16 = wq^T / wo^T (128x128), wk16/wv16 = wk^T / wv^T (32x128)
__global__ __launch_bounds__(256) void k_prep(
    const float* __restrict__ fcw, const float* __restrict__ prw,
    const float* __restrict__ wq, const float* __restrict__ wo,
    const float* __restrict__ wk, const float* __restrict__ wv,
    _Float16* __restrict__ fcw16, _Float16* __restrict__ prw16,
    _Float16* __restrict__ wq16, _Float16* __restrict__ wo16,
    _Float16* __restrict__ wk16, _Float16* __restrict__ wv16) {
  int t = blockIdx.x * 256 + threadIdx.x;
  if (t < 16384) { fcw16[t] = (_Float16)fcw[t]; return; }
  t -= 16384;
  if (t < 16384) { prw16[t] = (_Float16)prw[t]; return; }
  t -= 16384;
  if (t < 16384) { int k = t >> 7, n = t & 127; wq16[t] = (_Float16)wq[n * DP + k]; return; }
  t -= 16384;
  if (t < 16384) { int k = t >> 7, n = t & 127; wo16[t] = (_Float16)wo[n * DP + k]; return; }
  t -= 16384;
  if (t < 4096)  { int k = t >> 7, n = t & 127; wk16[t] = (_Float16)wk[n * DE + k]; return; }
  t -= 4096;
  if (t < 4096)  { int k = t >> 7, n = t & 127; wv16[t] = (_Float16)wv[n * DE + k]; return; }
}

// ---------------------------------------------------------------------------
// y1 = gelu(LN(y,ln4) @ fc_w + fc_b) @ pr_w + pr_b + y       (98304 x 128 rows)
__global__ __launch_bounds__(256) void k_mlp(
    const float* __restrict__ y, const float* __restrict__ ln4w, const float* __restrict__ ln4b,
    const _Float16* __restrict__ fcw16, const float* __restrict__ fcb,
    const _Float16* __restrict__ prw16, const float* __restrict__ prb,
    float* __restrict__ y1) {
  __shared__ _Float16 lds[8][16 * DP];
  const int lane = threadIdx.x & 31, wid = threadIdx.x >> 5;
  const int row0 = (blockIdx.x * 8 + wid) * 16;
  _Float16* tile = lds[wid];
  __builtin_prefetch(fcw16 + threadIdx.x * 64, 0, 1);
  __builtin_prefetch(prw16 + threadIdx.x * 64, 0, 1);
  // LayerNorm: lane handles row m=lane&15, half hh=lane>>4
  {
    int m = lane & 15, hh = lane >> 4;
    const float* r = y + (size_t)(row0 + m) * DP;
    float s = 0.f, s2 = 0.f;
#pragma unroll
    for (int i = 0; i < 64; ++i) { float v = r[hh * 64 + i]; s += v; s2 += v * v; }
    s += __shfl_xor(s, 16);
    s2 += __shfl_xor(s2, 16);
    float mu = s * (1.f / DP);
    float var = s2 * (1.f / DP) - mu * mu;
    float rinv = rsqrtf(var + 1e-5f);
#pragma unroll
    for (int i = 0; i < 64; ++i) {
      int c = hh * 64 + i;
      tile[m * DP + c] = (_Float16)((r[c] - mu) * rinv * ln4w[c] + ln4b[c]);
    }
  }
  __syncthreads();
  // GEMM1: all 8 N-tiles, K=128
  v8f acc[8];
#pragma unroll
  for (int nt = 0; nt < 8; ++nt) {
    v8f c = {};
#pragma unroll
    for (int kt = 0; kt < 4; ++kt) {
      v16h a = ldsA_frag(tile, DP, kt * 32, lane);
      v16h b = gB16(fcw16 + (size_t)(kt * 32) * DP + nt * 16, lane);
      c = WMMA_F16(a, b, c);
    }
    acc[nt] = c;
  }
  __syncthreads();
  // bias + gelu -> overwrite LDS with h (f16)
  {
    int n = lane & 15, g = lane >> 4;
#pragma unroll
    for (int nt = 0; nt < 8; ++nt) {
      int col = nt * 16 + n;
      float bias = fcb[col];
#pragma unroll
      for (int j = 0; j < 8; ++j) {
        int m = j + 8 * g;
        tile[m * DP + col] = (_Float16)gelu_f(acc[nt][j] + bias);
      }
    }
  }
  __syncthreads();
  // GEMM2: y1 = h @ pr_w + pr_b + y
  {
    int n = lane & 15, g = lane >> 4;
#pragma unroll
    for (int nt = 0; nt < 8; ++nt) {
      v8f c = {};
#pragma unroll
      for (int kt = 0; kt < 4; ++kt) {
        v16h a = ldsA_frag(tile, DP, kt * 32, lane);
        v16h b = gB16(prw16 + (size_t)(kt * 32) * DP + nt * 16, lane);
        c = WMMA_F16(a, b, c);
      }
      int col = nt * 16 + n;
      float bias = prb[col];
#pragma unroll
      for (int j = 0; j < 8; ++j) {
        int m = j + 8 * g;
        size_t o = (size_t)(row0 + m) * DP + col;
        y1[o] = c[j] + bias + y[o];
      }
    }
  }
}

// ---------------------------------------------------------------------------
// ctx = LN(x, ln2); k = heads(ctx @ wk^T); v = heads(ctx @ wv^T)   (51200 rows, K=32)
__global__ __launch_bounds__(256) void k_kv(
    const float* __restrict__ x, const float* __restrict__ ln2w, const float* __restrict__ ln2b,
    const _Float16* __restrict__ wk16, const _Float16* __restrict__ wv16,
    float* __restrict__ kbuf, _Float16* __restrict__ vbuf) {
  __shared__ _Float16 lds[8][16 * DE];
  const int lane = threadIdx.x & 31, wid = threadIdx.x >> 5;
  const int row0 = (blockIdx.x * 8 + wid) * 16;
  _Float16* tile = lds[wid];
  {
    int m = lane & 15, hh = lane >> 4;
    const float* r = x + (size_t)(row0 + m) * DE;
    float s = 0.f, s2 = 0.f;
#pragma unroll
    for (int i = 0; i < 16; ++i) { float v = r[hh * 16 + i]; s += v; s2 += v * v; }
    s += __shfl_xor(s, 16);
    s2 += __shfl_xor(s2, 16);
    float mu = s * (1.f / DE);
    float var = s2 * (1.f / DE) - mu * mu;
    float rinv = rsqrtf(var + 1e-5f);
#pragma unroll
    for (int i = 0; i < 16; ++i) {
      int c = hh * 16 + i;
      tile[m * DE + c] = (_Float16)((r[c] - mu) * rinv * ln2w[c] + ln2b[c]);
    }
  }
  __syncthreads();
  v16h a = ldsA_frag(tile, DE, 0, lane);
  int n = lane & 15, g = lane >> 4;
#pragma unroll
  for (int nt = 0; nt < 8; ++nt) {
    v16h bk = gB16(wk16 + nt * 16, lane);
    v16h bv = gB16(wv16 + nt * 16, lane);
    v8f ck = {}, cv = {};
    ck = WMMA_F16(a, bk, ck);
    cv = WMMA_F16(a, bv, cv);
    int col = nt * 16 + n, hd = col >> 4, d = col & 15;
#pragma unroll
    for (int j = 0; j < 8; ++j) {
      int rg = row0 + j + 8 * g;
      int bb = rg / NCTX, nn = rg % NCTX;
      size_t o = (((size_t)(bb * H + hd)) * NCTX + nn) * DH + d;
      kbuf[o] = ck[j];
      vbuf[o] = (_Float16)cv[j];
    }
  }
}

// ---------------------------------------------------------------------------
// q = heads(LN(y1, ln1) @ wq^T)      (98304 rows, K=128)
__global__ __launch_bounds__(256) void k_q(
    const float* __restrict__ y1, const float* __restrict__ ln1w, const float* __restrict__ ln1b,
    const _Float16* __restrict__ wq16, float* __restrict__ qbuf) {
  __shared__ _Float16 lds[8][16 * DP];
  const int lane = threadIdx.x & 31, wid = threadIdx.x >> 5;
  const int row0 = (blockIdx.x * 8 + wid) * 16;
  _Float16* tile = lds[wid];
  __builtin_prefetch(wq16 + threadIdx.x * 64, 0, 1);
  {
    int m = lane & 15, hh = lane >> 4;
    const float* r = y1 + (size_t)(row0 + m) * DP;
    float s = 0.f, s2 = 0.f;
#pragma unroll
    for (int i = 0; i < 64; ++i) { float v = r[hh * 64 + i]; s += v; s2 += v * v; }
    s += __shfl_xor(s, 16);
    s2 += __shfl_xor(s2, 16);
    float mu = s * (1.f / DP);
    float var = s2 * (1.f / DP) - mu * mu;
    float rinv = rsqrtf(var + 1e-5f);
#pragma unroll
    for (int i = 0; i < 64; ++i) {
      int c = hh * 64 + i;
      tile[m * DP + c] = (_Float16)((r[c] - mu) * rinv * ln1w[c] + ln1b[c]);
    }
  }
  __syncthreads();
  int n = lane & 15, g = lane >> 4;
#pragma unroll
  for (int nt = 0; nt < 8; ++nt) {
    v8f c = {};
#pragma unroll
    for (int kt = 0; kt < 4; ++kt) {
      v16h a = ldsA_frag(tile, DP, kt * 32, lane);
      v16h b = gB16(wq16 + (size_t)(kt * 32) * DP + nt * 16, lane);
      c = WMMA_F16(a, b, c);
    }
    int col = nt * 16 + n, hd = col >> 4, d = col & 15;
#pragma unroll
    for (int j = 0; j < 8; ++j) {
      int rg = row0 + j + 8 * g;
      int b2 = rg / NQ, nn = rg % NQ;
      qbuf[(((size_t)(b2 * H + hd)) * NQ + nn) * DH + d] = c[j];
    }
  }
}

// ---------------------------------------------------------------------------
// Build A fragment (16 rows x K=32, only first 16 K valid) from (n,16) rows,
// scaled by DN; also return per-lane sum of squares (for diag).
__device__ __forceinline__ v16h feat_A(const float* rows, int lane, float* sq_out) {
  int m = lane & 15, kh = lane >> 4;
  float sq = 0.f;
  v16h a;
#pragma unroll
  for (int j = 0; j < 8; ++j) {
    int k = a_kidx(j, kh);
    float v0 = (k < 16) ? rows[m * DH + k] : 0.f;
    float v1 = (k + 1 < 16) ? rows[m * DH + k + 1] : 0.f;
    sq += v0 * v0 + v1 * v1;
    a[2 * j]     = (_Float16)(DN * v0);
    a[2 * j + 1] = (_Float16)(DN * v1);
  }
  *sq_out = sq;
  return a;
}

__device__ __forceinline__ v16h proj_B(const float* proj, int ft, int lane) {
  int n = lane & 15, kh = lane >> 4;
  v16h b;
#pragma unroll
  for (int j = 0; j < 8; ++j) {
    int k = a_kidx(j, kh);
    b[2 * j]     = (k < 16) ? (_Float16)proj[(ft * 16 + n) * DH + k] : (_Float16)0.f;
    b[2 * j + 1] = (k + 1 < 16) ? (_Float16)proj[(ft * 16 + n) * DH + k + 1] : (_Float16)0.f;
  }
  return b;
}

// q features: dash via WMMA, per-row max, q_unnorm -> d_out, rowmax -> ws
__global__ __launch_bounds__(256) void k_qfeat(
    const float* __restrict__ qbuf, const float* __restrict__ proj,
    float* __restrict__ qun, float* __restrict__ qrowmax) {
  __shared__ float diagb[8][16];
  const int lane = threadIdx.x & 31, wid = threadIdx.x >> 5;
  const int task = blockIdx.x * 8 + wid;
  const int bh = task / 192, n0 = (task % 192) * 16;
  const float* rows = qbuf + (((size_t)bh) * NQ + n0) * DH;
  float sq;
  v16h a = feat_A(rows, lane, &sq);
  sq += __shfl_xor(sq, 16);
  if (lane < 16) diagb[wid][lane & 15] = DIAG_C * sq;
  v8f acc[4];
#pragma unroll
  for (int ft = 0; ft < 4; ++ft) {
    v16h b = proj_B(proj, ft, lane);
    v8f c = {};
    acc[ft] = WMMA_F16(a, b, c);
  }
  int g = lane >> 4, fcol = lane & 15;
#pragma unroll
  for (int j = 0; j < 8; ++j) {
    float r = fmaxf(fmaxf(acc[0][j], acc[1][j]), fmaxf(acc[2][j], acc[3][j]));
#pragma unroll
    for (int off = 1; off < 16; off <<= 1) r = fmaxf(r, __shfl_xor(r, off));
    if ((lane & 15) == 0) qrowmax[(size_t)bh * NQ + n0 + j + 8 * g] = r;
  }
#pragma unroll
  for (int ft = 0; ft < 4; ++ft) {
#pragma unroll
    for (int j = 0; j < 8; ++j) {
      int m = j + 8 * g;
      float un = RATIO * expf(acc[ft][j] - diagb[wid][m]);
      qun[(((size_t)bh) * NQ + n0 + m) * M + ft * 16 + fcol] = un;
    }
  }
}

// k feature pass 1: per-(b,h) global max of dash (atomic max)
__global__ __launch_bounds__(256) void k_kmax(
    const float* __restrict__ kbuf, const float* __restrict__ proj, float* __restrict__ kmax) {
  const int lane = threadIdx.x & 31, wid = threadIdx.x >> 5;
  const int task = blockIdx.x * 8 + wid;
  const int bh = task / 192, n0 = (task % 192) * 16;
  const int b2 = bh >> 3, h = bh & 7, v = b2 >> 4, b = b2 & 15;
  const float* rows = kbuf + (((size_t)(b * H + h)) * NCTX + n0 + v * S) * DH;
  float sq;
  v16h a = feat_A(rows, lane, &sq);
  float tmax = -__builtin_inff();
#pragma unroll
  for (int ft = 0; ft < 4; ++ft) {
    v16h bf = proj_B(proj, ft, lane);
    v8f c = {};
    c = WMMA_F16(a, bf, c);
#pragma unroll
    for (int j = 0; j < 8; ++j) tmax = fmaxf(tmax, c[j]);
  }
#pragma unroll
  for (int off = 1; off < 32; off <<= 1) tmax = fmaxf(tmax, __shfl_xor(tmax, off));
  if (lane == 0) atomicMaxF(&kmax[bh], tmax);
}

// k feature pass 2: k_unnorm -> d_out; accumulate k_sum = sum_n kf (atomics)
__global__ __launch_bounds__(256) void k_kfeat(
    const float* __restrict__ kbuf, const float* __restrict__ proj,
    const float* __restrict__ kmax, float* __restrict__ kun, float* __restrict__ ksum) {
  __shared__ float diagb[8][16];
  const int lane = threadIdx.x & 31, wid = threadIdx.x >> 5;
  const int task = blockIdx.x * 8 + wid;
  const int bh = task / 192, n0 = (task % 192) * 16;
  const int b2 = bh >> 3, h = bh & 7, v = b2 >> 4, b = b2 & 15;
  const float* rows = kbuf + (((size_t)(b * H + h)) * NCTX + n0 + v * S) * DH;
  float sq;
  v16h a = feat_A(rows, lane, &sq);
  sq += __shfl_xor(sq, 16);
  if (lane < 16) diagb[wid][lane & 15] = DIAG_C * sq;
  const float esc = expf(-kmax[bh]);
  int g = lane >> 4, fcol = lane & 15;
#pragma unroll
  for (int ft = 0; ft < 4; ++ft) {
    v16h bf = proj_B(proj, ft, lane);
    v8f c = {};
    c = WMMA_F16(a, bf, c);
    float colsum = 0.f;
#pragma unroll
    for (int j = 0; j < 8; ++j) {
      int m = j + 8 * g;
      float un = RATIO * expf(c[j] - diagb[wid][m]);
      kun[(((size_t)bh) * NQ + n0 + m) * M + ft * 16 + fcol] = un;
      colsum += un * esc + REPS;
    }
    colsum += __shfl_xor(colsum, 16);
    if (lane < 16) atomicAdd(&ksum[(size_t)bh * M + ft * 16 + lane], colsum);
  }
}

// ---------------------------------------------------------------------------
// ctx2[bh][f][e] = sum_n kf[n][f] * v2[n][e]   (K = 3072, split across 8 waves)
__global__ __launch_bounds__(256) void k_ctx(
    const float* __restrict__ kun, const float* __restrict__ kmax,
    const _Float16* __restrict__ vbuf, float* __restrict__ ctx2) {
  const int lane = threadIdx.x & 31, wid = threadIdx.x >> 5;
  const int bh = blockIdx.x >> 2, ft = blockIdx.x & 3;
  const int b2 = bh >> 3, h = bh & 7, v = b2 >> 4, b = b2 & 15;
  const float esc = expf(-kmax[bh]);
  const float* kub = kun + (size_t)bh * NQ * M;
  const _Float16* vr0 = vbuf + (((size_t)(b * H + h)) * NCTX + v * S) * DH;
  const int mf = lane & 15, kh = lane >> 4;
  const int f = ft * 16 + mf;
  v8f acc = {};
  for (int s = 0; s < 12; ++s) {
    const int nb = wid * 384 + s * 32;
    v16h a;
#pragma unroll
    for (int j = 0; j < 8; ++j) {
      int k = a_kidx(j, kh);
      a[2 * j]     = (_Float16)(kub[(size_t)(nb + k) * M + f] * esc + REPS);
      a[2 * j + 1] = (_Float16)(kub[(size_t)(nb + k + 1) * M + f] * esc + REPS);
    }
    v16h bf;
#pragma unroll
    for (int j = 0; j < 8; ++j) {
      int k = a_kidx(j, kh);
      bf[2 * j]     = vr0[(size_t)(nb + k) * DH + mf];
      bf[2 * j + 1] = vr0[(size_t)(nb + k + 1) * DH + mf];
    }
    acc = WMMA_F16(a, bf, acc);
  }
  const int e = lane & 15, g = lane >> 4;
#pragma unroll
  for (int j = 0; j < 8; ++j) {
    int ff = ft * 16 + j + 8 * g;
    atomicAdd(&ctx2[(size_t)bh * M * DH + ff * DH + e], acc[j]);
  }
}

// ---------------------------------------------------------------------------
// Final: att = d_inv * (qf @ ctx2) per head; out = att @ wo^T + wo_b + y1
__global__ __launch_bounds__(256) void k_final(
    const float* __restrict__ qun, const float* __restrict__ qrowmax,
    const float* __restrict__ ksum, const float* __restrict__ ctx2,
    const _Float16* __restrict__ wo16, const float* __restrict__ wob,
    const float* __restrict__ y1, float* __restrict__ yout) {
  __shared__ _Float16 attl[8][16 * DP];
  __shared__ float dinvb[8][16];
  const int lane = threadIdx.x & 31, wid = threadIdx.x >> 5;
  const int task = blockIdx.x * 8 + wid;
  const int b2 = task / 192, n0 = (task % 192) * 16;
  _Float16* att = attl[wid];
  __builtin_prefetch(wo16 + threadIdx.x * 64, 0, 1);
  const int mrow = lane & 15, hh = lane >> 4, ncol = lane & 15;
  for (int h = 0; h < H; ++h) {
    const int bh = b2 * H + h;
    const float* qub = qun + (size_t)bh * NQ * M;
    const float* qrm = qrowmax + (size_t)bh * NQ;
    const int r = n0 + mrow;
    const float qs = expf(-qrm[r]);
    const float* qr = qub + (size_t)r * M;
    // d_inv = 1 / (qf . k_sum) per row (two lanes per row)
    {
      const float* ks = ksum + (size_t)bh * M;
      float sacc = 0.f;
#pragma unroll
      for (int i = 0; i < 32; ++i) {
        int f = hh * 32 + i;
        sacc += (qr[f] * qs + REPS) * ks[f];
      }
      sacc += __shfl_xor(sacc, 16);
      if (lane < 16) dinvb[wid][mrow] = 1.f / sacc;
    }
    // att tile = qf(16x64) @ ctx2(64x16)
    v8f c = {};
#pragma unroll
    for (int kt = 0; kt < 2; ++kt) {
      v16h a;
#pragma unroll
      for (int j = 0; j < 8; ++j) {
        int f = kt * 32 + a_kidx(j, hh);
        a[2 * j]     = (_Float16)(qr[f] * qs + REPS);
        a[2 * j + 1] = (_Float16)(qr[f + 1] * qs + REPS);
      }
      const float* cb = ctx2 + (size_t)bh * M * DH + (size_t)(kt * 32) * DH + ncol;
      v16h bf;
#pragma unroll
      for (int j = 0; j < 8; ++j) {
        int k = a_kidx(j, hh);
        bf[2 * j]     = (_Float16)cb[k * DH];
        bf[2 * j + 1] = (_Float16)cb[(k + 1) * DH];
      }
      c = WMMA_F16(a, bf, c);
    }
#pragma unroll
    for (int j = 0; j < 8; ++j) {
      int m = j + 8 * hh;
      att[m * DP + h * DH + ncol] = (_Float16)(c[j] * dinvb[wid][m]);
    }
  }
  __syncthreads();
  // out = att @ wo^T + wo_b + y1
#pragma unroll
  for (int nt = 0; nt < 8; ++nt) {
    v8f c = {};
#pragma unroll
    for (int kt = 0; kt < 4; ++kt) {
      v16h a = ldsA_frag(att, DP, kt * 32, lane);
      v16h b = gB16(wo16 + (size_t)(kt * 32) * DP + nt * 16, lane);
      c = WMMA_F16(a, b, c);
    }
    int col = nt * 16 + ncol;
    float bias = wob[col];
#pragma unroll
    for (int j = 0; j < 8; ++j) {
      int m = j + 8 * hh;
      size_t rr = (size_t)b2 * NQ + n0 + m;
      yout[rr * DP + col] = c[j] + bias + y1[rr * DP + col];
    }
  }
}

// ---------------------------------------------------------------------------
extern "C" void kernel_launch(void* const* d_in, const int* in_sizes, int n_in,
                              void* d_out, int out_size, void* d_ws, size_t ws_size,
                              hipStream_t stream) {
  (void)in_sizes; (void)n_in; (void)out_size; (void)ws_size;
  const float* x    = (const float*)d_in[0];
  const float* y    = (const float*)d_in[1];
  const float* proj = (const float*)d_in[2];
  const float* ln1w = (const float*)d_in[3];
  const float* ln1b = (const float*)d_in[4];
  const float* ln2w = (const float*)d_in[5];
  const float* ln2b = (const float*)d_in[6];
  const float* ln4w = (const float*)d_in[7];
  const float* ln4b = (const float*)d_in[8];
  const float* fcw  = (const float*)d_in[9];
  const float* fcb  = (const float*)d_in[10];
  const float* prw  = (const float*)d_in[11];
  const float* prb  = (const float*)d_in[12];
  const float* wq   = (const float*)d_in[13];
  const float* wk   = (const float*)d_in[14];
  const float* wv   = (const float*)d_in[15];
  const float* wo   = (const float*)d_in[16];
  const float* wob  = (const float*)d_in[17];

  float* out  = (float*)d_out;
  float* yout = out;                                  // (2,B,NQ,DP)
  float* qun  = yout + (size_t)B2 * NQ * DP;          // (2B,H,1,NQ,M)
  float* kun  = qun + (size_t)BH * NQ * M;            // (2B,H,1,NQ,M)

  char* w = (char*)d_ws;
  float*     y1      = (float*)w;     w += (size_t)B2 * NQ * DP * 4;        // 50.3 MB
  float*     kbuf    = (float*)w;     w += (size_t)Bb * H * NCTX * DH * 4;  // 26.2 MB
  _Float16*  vbuf    = (_Float16*)w;  w += (size_t)Bb * H * NCTX * DH * 2;  // 13.1 MB
  float*     qbuf    = (float*)w;     w += (size_t)BH * NQ * DH * 4;        // 50.3 MB
  float*     qrowmax = (float*)w;     w += (size_t)BH * NQ * 4;             // 3.1 MB
  float*     kmax    = (float*)w;     w += 1024;
  float*     ksum    = (float*)w;     w += (size_t)BH * M * 4;              // 64 KB
  float*     ctx2    = (float*)w;     w += (size_t)BH * M * DH * 4;         // 1 MB
  _Float16*  fcw16   = (_Float16*)w;  w += 16384 * 2;
  _Float16*  prw16   = (_Float16*)w;  w += 16384 * 2;
  _Float16*  wq16    = (_Float16*)w;  w += 16384 * 2;
  _Float16*  wo16    = (_Float16*)w;  w += 16384 * 2;
  _Float16*  wk16    = (_Float16*)w;  w += 4096 * 2;
  _Float16*  wv16    = (_Float16*)w;  w += 4096 * 2;

  // zero k_sum + ctx2 (contiguous), init kmax to -inf, convert weights to f16
  hipMemsetAsync(ksum, 0, (size_t)BH * M * 4 + (size_t)BH * M * DH * 4, stream);
  k_init<<<1, 256, 0, stream>>>(kmax);
  k_prep <<<288, 256, 0, stream>>>(fcw, prw, wq, wo, wk, wv,
                                   fcw16, prw16, wq16, wo16, wk16, wv16);

  k_mlp  <<<768, 256, 0, stream>>>(y, ln4w, ln4b, fcw16, fcb, prw16, prb, y1);
  k_kv   <<<400, 256, 0, stream>>>(x, ln2w, ln2b, wk16, wv16, kbuf, vbuf);
  k_q    <<<768, 256, 0, stream>>>(y1, ln1w, ln1b, wq16, qbuf);
  k_qfeat<<<6144, 256, 0, stream>>>(qbuf, proj, qun, qrowmax);
  k_kmax <<<6144, 256, 0, stream>>>(kbuf, proj, kmax);
  k_kfeat<<<6144, 256, 0, stream>>>(kbuf, proj, kmax, kun, ksum);
  k_ctx  <<<1024, 256, 0, stream>>>(kun, kmax, vbuf, ctx2);
  k_final<<<768, 256, 0, stream>>>(qun, qrowmax, ksum, ctx2, wo16, wob, y1, yout);
}
